// GCN_74483322847349
// MI455X (gfx1250) — compile-verified, matrix operand order checked by reading, not run on hardware
//
#include <hip/hip_runtime.h>

#define N_NODES  100000
#define N_EDGES  200000
#define N_GRAPHS 512
#define EMB_DIM  256
#define HID_DIM  128

typedef __attribute__((ext_vector_type(16))) __bf16 v16bf;
typedef __attribute__((ext_vector_type(8)))  float  v8f;

union Frag16 { v16bf v; unsigned u[8]; uint4 q[2]; };

__device__ __forceinline__ unsigned short f2bf(float x) {
  unsigned b = __float_as_uint(x);
  b += 0x7FFFu + ((b >> 16) & 1u);      // round-to-nearest-even
  return (unsigned short)(b >> 16);
}

// ---------------- degree / dinv ----------------
__global__ void k_deg(const int* __restrict__ dst, float* __restrict__ deg, int E) {
  int e = blockIdx.x * blockDim.x + threadIdx.x;
  if (e < E) atomicAdd(&deg[dst[e]], 1.0f);
}
__global__ void k_dinv(float* __restrict__ dinv, int N) {
  int i = blockIdx.x * blockDim.x + threadIdx.x;
  if (i < N) dinv[i] = rsqrtf(dinv[i] + 1.0f);   // +1 self loop; always > 0
}

// ---------------- embedding gather -> bf16 ----------------
__global__ void k_embed(const int* __restrict__ x, const float* __restrict__ emb,
                        unsigned short* __restrict__ h0, int total) {
  int t = blockIdx.x * blockDim.x + threadIdx.x;
  if (t >= total) return;
  int i = t >> 8, f = t & 255;
  h0[t] = f2bf(emb[(size_t)x[i] * EMB_DIM + f]);
}

// ---------------- pack weight [K,128] f32 into WMMA B-fragment layout ----------------
__global__ void k_prepack(const float* __restrict__ W, unsigned* __restrict__ Bp, int K) {
  int t = blockIdx.x * blockDim.x + threadIdx.x;
  int total = (K >> 5) << 11;                    // (K/32) * 8 colTiles * 32 lanes * 8 pairs
  if (t >= total) return;
  int j = t & 7, lane = (t >> 3) & 31, ct = (t >> 8) & 7, kt = t >> 11;
  int hi = lane >> 4;
  int k = kt * 32 + hi * 8 + ((j >> 2) << 4) + ((j & 3) << 1);
  int n = ct * 16 + (lane & 15);
  unsigned lo  = f2bf(W[(size_t)k       * HID_DIM + n]);
  unsigned hiw = f2bf(W[(size_t)(k + 1) * HID_DIM + n]);
  Bp[t] = lo | (hiw << 16);
}

// ---------------- GEMM: C[N,128] = A[N,K](bf16) @ Wpack (WMMA bf16) ----------------
template<int K>
__global__ __launch_bounds__(256) void k_gemm(const unsigned short* __restrict__ A,
                                              const unsigned* __restrict__ Bp,
                                              float* __restrict__ C) {
  const int rowBase = blockIdx.x << 4;           // 16 rows per block
  const int wave = threadIdx.x >> 5;             // col tile 0..7
  const int lane = threadIdx.x & 31;
  const int hi   = lane >> 4;
  const int mrow = rowBase + (lane & 15);
  const unsigned short* arow = A + (size_t)mrow * K + hi * 8;
  const unsigned* bp = Bp + ((size_t)wave * 32 + lane) * 8;
  v8f acc = {0.f, 0.f, 0.f, 0.f, 0.f, 0.f, 0.f, 0.f};
#pragma unroll
  for (int kt = 0; kt < (K >> 5); ++kt) {
    Frag16 a, b;
    const uint4* ap = (const uint4*)(arow + kt * 32);
    a.q[0] = ap[0];                              // K pairs 0,2,4,6   (+hi*8)
    a.q[1] = ap[2];                              // K pairs 16,18,20,22 (+hi*8)
    const uint4* bq = (const uint4*)(bp + (size_t)kt * 2048);
    b.q[0] = bq[0];
    b.q[1] = bq[1];
    acc = __builtin_amdgcn_wmma_f32_16x16x32_bf16(false, a.v, false, b.v,
                                                  (short)0, acc, false, false);
  }
  int n = (wave << 4) + (lane & 15);
#pragma unroll
  for (int r = 0; r < 8; ++r)
    C[(size_t)(rowBase + r + hi * 8) * HID_DIM + n] = acc[r];
}

// ---------------- edge aggregation: acc[dst] += dinv[s]*dinv[d]*lin[src] ----------------
__global__ void k_agg(const int* __restrict__ src, const int* __restrict__ dst,
                      const float* __restrict__ dinv, const float* __restrict__ lin,
                      float* __restrict__ acc, int E) {
  int t = blockIdx.x * blockDim.x + threadIdx.x;
  int e = t >> 5;
  if (e >= E) return;
  int lane = t & 31;
  int s = src[e], d = dst[e];
  float nrm = dinv[s] * dinv[d];
  float4 v = *((const float4*)(lin + (size_t)s * HID_DIM) + lane);
  float* out = acc + (size_t)d * HID_DIM + lane * 4;
  atomicAdd(out + 0, nrm * v.x);
  atomicAdd(out + 1, nrm * v.y);
  atomicAdd(out + 2, nrm * v.z);
  atomicAdd(out + 3, nrm * v.w);
}

// ---------------- finalize: self-loop + bias + relu ----------------
__global__ void k_fin_bf16(const float* __restrict__ acc, const float* __restrict__ lin,
                           const float* __restrict__ dinv, const float* __restrict__ bias,
                           unsigned short* __restrict__ out, int total) {
  int t = blockIdx.x * blockDim.x + threadIdx.x;
  if (t >= total) return;
  int i = t >> 7, f = t & 127;
  float dv = dinv[i];
  float v = acc[t] + dv * dv * lin[t] + bias[f];
  out[t] = f2bf(fmaxf(v, 0.f));
}
__global__ void k_fin_f32(const float* __restrict__ acc, const float* __restrict__ lin,
                          const float* __restrict__ dinv, const float* __restrict__ bias,
                          float* __restrict__ out, int total) {
  int t = blockIdx.x * blockDim.x + threadIdx.x;
  if (t >= total) return;
  int i = t >> 7, f = t & 127;
  float dv = dinv[i];
  out[t] = fmaxf(acc[t] + dv * dv * lin[t] + bias[f], 0.f);
}

// ---------------- attention pool ----------------
__global__ void k_gate(const float* __restrict__ h, const float* __restrict__ gW,
                       const float* __restrict__ gb, const int* __restrict__ batch,
                       float* __restrict__ gate, unsigned* __restrict__ menc, int N) {
  int i = blockIdx.x * blockDim.x + threadIdx.x;
  if (i >= N) return;
  const float* hr = h + (size_t)i * HID_DIM;
  float g = gb[0];
  for (int f = 0; f < HID_DIM; ++f) g += hr[f] * gW[f];
  gate[i] = g;
  unsigned b = __float_as_uint(g);
  unsigned key = (b & 0x80000000u) ? ~b : (b | 0x80000000u);  // order-preserving
  atomicMax(&menc[batch[i]], key);
}
__global__ void k_exp(const float* __restrict__ gate, const unsigned* __restrict__ menc,
                      const int* __restrict__ batch, float* __restrict__ ebuf,
                      float* __restrict__ denom, int N) {
  int i = blockIdx.x * blockDim.x + threadIdx.x;
  if (i >= N) return;
  int b = batch[i];
  unsigned k = menc[b];
  float m = __uint_as_float((k & 0x80000000u) ? (k & 0x7FFFFFFFu) : ~k);
  float e = expf(gate[i] - m);
  ebuf[i] = e;
  atomicAdd(&denom[b], e);
}
__global__ void k_pool(const float* __restrict__ h, const float* __restrict__ ebuf,
                       const float* __restrict__ denom, const int* __restrict__ batch,
                       float* __restrict__ pooled, int N) {
  int t = blockIdx.x * blockDim.x + threadIdx.x;
  int i = t >> 5;
  if (i >= N) return;
  int lane = t & 31;
  int b = batch[i];
  float alpha = ebuf[i] / denom[b];
  float4 v = *((const float4*)(h + (size_t)i * HID_DIM) + lane);
  float* out = pooled + (size_t)b * HID_DIM + lane * 4;
  atomicAdd(out + 0, alpha * v.x);
  atomicAdd(out + 1, alpha * v.y);
  atomicAdd(out + 2, alpha * v.z);
  atomicAdd(out + 3, alpha * v.w);
}
__global__ void k_out(const float* __restrict__ pooled, const float* __restrict__ cW,
                      const float* __restrict__ cb, float* __restrict__ out, int G) {
  int g = blockIdx.x * blockDim.x + threadIdx.x;
  if (g >= G) return;
  const float* p = pooled + (size_t)g * HID_DIM;
  float o = cb[0];
  for (int f = 0; f < HID_DIM; ++f) o += p[f] * cW[f];
  out[g] = o;
}

extern "C" void kernel_launch(void* const* d_in, const int* in_sizes, int n_in,
                              void* d_out, int out_size, void* d_ws, size_t ws_size,
                              hipStream_t stream) {
  const int*   x     = (const int*)d_in[0];
  const int*   eidx  = (const int*)d_in[1];
  const int*   batch = (const int*)d_in[2];
  const float* emb   = (const float*)d_in[3];
  const float* W1    = (const float*)d_in[4];
  const float* b1    = (const float*)d_in[5];
  const float* W2    = (const float*)d_in[6];
  const float* b2    = (const float*)d_in[7];
  const float* gW    = (const float*)d_in[8];
  const float* gb    = (const float*)d_in[9];
  const float* cW    = (const float*)d_in[10];
  const float* cb    = (const float*)d_in[11];
  const int* src = eidx;
  const int* dst = eidx + N_EDGES;
  float* out = (float*)d_out;

  char* ws = (char*)d_ws;
  size_t off = 0;
  auto carve = [&](size_t bytes) { void* p = ws + off; off = (off + bytes + 255) & ~(size_t)255; return p; };
  unsigned short* h0bf   = (unsigned short*)carve((size_t)N_NODES * EMB_DIM * 2); // later reused as h2 (f32, same size)
  float*          lin    = (float*)carve((size_t)N_NODES * HID_DIM * 4);
  float*          acc    = (float*)carve((size_t)N_NODES * HID_DIM * 4);
  unsigned short* h1bf   = (unsigned short*)carve((size_t)N_NODES * HID_DIM * 2);
  float*          dinv   = (float*)carve((size_t)N_NODES * 4);
  float*          gate   = (float*)carve((size_t)N_NODES * 4);
  float*          ebuf   = (float*)carve((size_t)N_NODES * 4);
  unsigned*       w1pack = (unsigned*)carve((size_t)(EMB_DIM / 32) * 2048 * 4);
  unsigned*       w2pack = (unsigned*)carve((size_t)(HID_DIM / 32) * 2048 * 4);
  unsigned*       menc   = (unsigned*)carve((size_t)N_GRAPHS * 4);
  float*          denom  = (float*)carve((size_t)N_GRAPHS * 4);
  float*          pooled = (float*)carve((size_t)N_GRAPHS * HID_DIM * 4);
  float*          h2     = (float*)h0bf;   // reuse: N*128*4 == N*256*2 bytes

  // degrees -> dinv (deg accumulated in-place in dinv buffer)
  hipMemsetAsync(dinv, 0, (size_t)N_NODES * 4, stream);
  k_deg<<<(N_EDGES + 255) / 256, 256, 0, stream>>>(dst, dinv, N_EDGES);
  k_dinv<<<(N_NODES + 255) / 256, 256, 0, stream>>>(dinv, N_NODES);

  // embedding -> bf16 ; weight prepack
  k_embed<<<(N_NODES * EMB_DIM) / 256, 256, 0, stream>>>(x, emb, h0bf, N_NODES * EMB_DIM);
  k_prepack<<<(EMB_DIM / 32) * 2048 / 256, 256, 0, stream>>>(W1, w1pack, EMB_DIM);
  k_prepack<<<(HID_DIM / 32) * 2048 / 256, 256, 0, stream>>>(W2, w2pack, HID_DIM);

  // conv1: linear (WMMA) -> scatter agg -> self loop + bias + relu -> bf16
  k_gemm<EMB_DIM><<<N_NODES / 16, 256, 0, stream>>>(h0bf, w1pack, lin);
  hipMemsetAsync(acc, 0, (size_t)N_NODES * HID_DIM * 4, stream);
  k_agg<<<(N_EDGES * 32) / 256, 256, 0, stream>>>(src, dst, dinv, lin, acc, N_EDGES);
  k_fin_bf16<<<(N_NODES * HID_DIM) / 256, 256, 0, stream>>>(acc, lin, dinv, b1, h1bf, N_NODES * HID_DIM);

  // conv2
  k_gemm<HID_DIM><<<N_NODES / 16, 256, 0, stream>>>(h1bf, w2pack, lin);
  hipMemsetAsync(acc, 0, (size_t)N_NODES * HID_DIM * 4, stream);
  k_agg<<<(N_EDGES * 32) / 256, 256, 0, stream>>>(src, dst, dinv, lin, acc, N_EDGES);
  k_fin_f32<<<(N_NODES * HID_DIM) / 256, 256, 0, stream>>>(acc, lin, dinv, b2, h2, N_NODES * HID_DIM);

  // attention pooling (segment softmax) + classifier
  hipMemsetAsync(menc, 0, (size_t)N_GRAPHS * 4, stream);    // 0 == encoded -NaN sentinel (min key)
  hipMemsetAsync(denom, 0, (size_t)N_GRAPHS * 4, stream);
  hipMemsetAsync(pooled, 0, (size_t)N_GRAPHS * HID_DIM * 4, stream);
  k_gate<<<(N_NODES + 255) / 256, 256, 0, stream>>>(h2, gW, gb, batch, gate, menc, N_NODES);
  k_exp<<<(N_NODES + 255) / 256, 256, 0, stream>>>(gate, menc, batch, ebuf, denom, N_NODES);
  k_pool<<<(N_NODES * 32) / 256, 256, 0, stream>>>(h2, ebuf, denom, batch, pooled, N_NODES);
  k_out<<<(N_GRAPHS + 255) / 256, 256, 0, stream>>>(pooled, cW, cb, out, N_GRAPHS);
}
